// mLSTMCell_24017457119634
// MI455X (gfx1250) — matrix operand order repaired and checked by reading, not compiled, not run on hardware
//
#include <hip/hip_runtime.h>
#include <math.h>

#define S_  2048
#define E_  1024
#define NH_ 8
#define DH_ 128
#define BM_ 64   // query rows per block (4 waves x 16)
#define BN_ 32   // key cols per tile
#define NT_ (S_ / BN_)  // 64 key tiles per head

typedef __attribute__((ext_vector_type(16))) __bf16 v16bf;
typedef __attribute__((ext_vector_type(8)))  float  v8f;
typedef __attribute__((ext_vector_type(4)))  int    v4i;

union U16 {
    unsigned short s[16];
    uint4          q[2];
    v16bf          v;
};

static __device__ __forceinline__ unsigned short f2bf(float f) {
    unsigned int u = __float_as_uint(f);
    u += 0x7FFFu + ((u >> 16) & 1u);   // round-to-nearest-even
    return (unsigned short)(u >> 16);
}

// 16 contiguous bf16 -> fragment
static __device__ __forceinline__ v16bf sfrag(const unsigned short* p) {
    U16 u;
    u.q[0] = *(const uint4*)(p);
    u.q[1] = *(const uint4*)(p + 8);
    return u.v;
}
// two 8-element bf16 chunks -> fragment (A-layout K split)
static __device__ __forceinline__ v16bf sfrag2(const unsigned short* p0,
                                               const unsigned short* p1) {
    U16 u;
    u.q[0] = *(const uint4*)(p0);
    u.q[1] = *(const uint4*)(p1);
    return u.v;
}

// ---------- DPP half-wave (16-lane) butterfly reduction ----------
constexpr int dpp8xor(int m) {
    int r = 0;
    for (int i = 0; i < 8; ++i) r |= ((i ^ m) & 7) << (3 * i);
    return r;
}
template <int IMM>
static __device__ __forceinline__ float dpp8f(float x) {
    return __int_as_float(__builtin_amdgcn_mov_dpp8(__float_as_int(x), IMM));
}
static __device__ __forceinline__ float dppror8f(float x) {
    int xi = __float_as_int(x);
    // DPP16 ROW_ROR:8 (0x128): lane i <- lane (i^8) within each 16-lane row
    return __int_as_float(__builtin_amdgcn_update_dpp(xi, xi, 0x128, 0xf, 0xf, false));
}
static __device__ __forceinline__ float hsum16(float x) {
    x += dpp8f<dpp8xor(1)>(x);
    x += dpp8f<dpp8xor(2)>(x);
    x += dpp8f<dpp8xor(4)>(x);
    x += dppror8f(x);
    return x;
}

// ---------- async global->LDS 16B copy (gfx1250) ----------
#if defined(__has_builtin)
#if __has_builtin(__builtin_amdgcn_global_load_async_to_lds_b128)
#define HAVE_ASYNC_BUILTIN 1
#endif
#if __has_builtin(__builtin_amdgcn_s_wait_asynccnt)
#define HAVE_WAITASYNC_BUILTIN 1
#endif
#endif

typedef __attribute__((address_space(1))) void* gas1_ptr;
typedef __attribute__((address_space(3))) void* las3_ptr;
typedef __attribute__((address_space(1))) v4i*  gv4_ptr;
typedef __attribute__((address_space(3))) v4i*  lv4_ptr;

static __device__ __forceinline__ void async_cp16(const unsigned short* g,
                                                  unsigned short* l) {
#if defined(HAVE_ASYNC_BUILTIN)
    __builtin_amdgcn_global_load_async_to_lds_b128(
        (gv4_ptr)(gas1_ptr)(void*)g, (lv4_ptr)(las3_ptr)(void*)l, 0, 0);
#else
    unsigned lofs = (unsigned)(unsigned long long)
        (__attribute__((address_space(3))) unsigned short*)l;
    asm volatile("global_load_async_to_lds_b128 %0, %1, off"
                 :
                 : "v"(lofs), "v"((unsigned long long)g)
                 : "memory");
#endif
}
// wait for all but the newest N async ops (N = one tile's worth for double buffer)
static __device__ __forceinline__ void wait_async0() {
#if defined(HAVE_WAITASYNC_BUILTIN)
    __builtin_amdgcn_s_wait_asynccnt(0);
#else
    asm volatile("s_wait_asynccnt 0x0" ::: "memory");
#endif
}
static __device__ __forceinline__ void wait_async8() {
#if defined(HAVE_WAITASYNC_BUILTIN)
    __builtin_amdgcn_s_wait_asynccnt(8);
#else
    asm volatile("s_wait_asynccnt 0x8" ::: "memory");
#endif
}

// -------------------- one-time bf16 conversion --------------------
// qb[t][e]     = bf16(q * 1/sqrt(DH))   (scale folded)
// kb[t][e]     = bf16(k)
// vtb[h][d][s] = bf16(v[s][h*DH+d])     (per-head transposed)
__global__ void mlstm_convert_kernel(const float* __restrict__ q,
                                     const float* __restrict__ k,
                                     const float* __restrict__ v,
                                     unsigned short* __restrict__ qb,
                                     unsigned short* __restrict__ kb,
                                     unsigned short* __restrict__ vtb) {
    const float invs = 0.08838834764831845f;  // 1/sqrt(128)
    const size_t i = (size_t)blockIdx.x * 256 + threadIdx.x;  // < S*E
    qb[i] = f2bf(q[i] * invs);
    kb[i] = f2bf(k[i]);
    const int h = (int)(i >> 18);          // / (DH*S)
    const int rem = (int)(i & 262143);
    const int d = rem >> 11;               // / S
    const int s = rem & 2047;
    vtb[i] = f2bf(v[(size_t)s * E_ + h * DH_ + d]);
}

// -------------------- gate pre-activations --------------------
__global__ void mlstm_gates_kernel(const float* __restrict__ q,
                                   const float* __restrict__ k,
                                   const float* __restrict__ v,
                                   const float* __restrict__ igw,
                                   const float* __restrict__ igb,
                                   const float* __restrict__ fgw,
                                   const float* __restrict__ fgb,
                                   float* __restrict__ ig,
                                   float* __restrict__ lsig) {
    const int t   = blockIdx.x;
    const int tid = threadIdx.x;
    float ai[NH_] = {0.f, 0.f, 0.f, 0.f, 0.f, 0.f, 0.f, 0.f};
    float af[NH_] = {0.f, 0.f, 0.f, 0.f, 0.f, 0.f, 0.f, 0.f};
    for (int c = tid; c < 3 * E_; c += 256) {
        float x = (c < E_) ? q[(size_t)t * E_ + c]
                 : (c < 2 * E_) ? k[(size_t)t * E_ + c - E_]
                                : v[(size_t)t * E_ + c - 2 * E_];
#pragma unroll
        for (int h = 0; h < NH_; ++h) {
            ai[h] = fmaf(x, igw[h * 3 * E_ + c], ai[h]);
            af[h] = fmaf(x, fgw[h * 3 * E_ + c], af[h]);
        }
    }
    __shared__ float red[16][256];
#pragma unroll
    for (int h = 0; h < NH_; ++h) {
        red[h][tid]       = ai[h];
        red[NH_ + h][tid] = af[h];
    }
    __syncthreads();
    if (tid < 16) {
        float s = 0.f;
        for (int i = 0; i < 256; ++i) s += red[tid][i];
        if (tid < NH_) {
            ig[tid * S_ + t] = s + igb[tid];
        } else {
            float x = s + fgb[tid - NH_];
            lsig[(tid - NH_) * S_ + t] = fminf(x, 0.f) - log1pf(expf(-fabsf(x)));
        }
    }
}

// ------------- cumsum + decay-max precompute (per head) -------------
// cs[t]   = cumsum(log_sigmoid(f))
// g[t]    = ig[t] - cs[t]
// pm[t]   = prefix max of g within t's 32-block
// tmax[b] = max of g over 32-block b
__global__ void mlstm_cumsum_kernel(const float* __restrict__ lsig,
                                    const float* __restrict__ ig,
                                    float* __restrict__ cs,
                                    float* __restrict__ g,
                                    float* __restrict__ pm,
                                    float* __restrict__ tmax) {
    const int h   = blockIdx.x;
    const int tid = threadIdx.x;
    __shared__ float tot[256];
    __shared__ float gb[S_];
    float p[8];
    float run = 0.f;
#pragma unroll
    for (int j = 0; j < 8; ++j) {
        run += lsig[h * S_ + tid * 8 + j];
        p[j] = run;
    }
    tot[tid] = run;
    __syncthreads();
    if (tid == 0) {
        float r = 0.f;
        for (int i = 0; i < 256; ++i) { float x = tot[i]; tot[i] = r; r += x; }
    }
    __syncthreads();
    const float off = tot[tid];
#pragma unroll
    for (int j = 0; j < 8; ++j) {
        const int t = tid * 8 + j;
        const float c = p[j] + off;
        cs[h * S_ + t] = c;
        const float gv = ig[h * S_ + t] - c;
        g[h * S_ + t] = gv;
        gb[t] = gv;
    }
    __syncthreads();
    if (tid < NT_) {
        float mmax = -INFINITY;
        for (int i = 0; i < BN_; ++i) {
            const int t = tid * BN_ + i;
            mmax = fmaxf(mmax, gb[t]);
            pm[h * S_ + t] = mmax;
        }
        tmax[h * NT_ + tid] = mmax;
    }
}

// -------------------- main flash-mLSTM kernel --------------------
__global__ void __launch_bounds__(128)
mlstm_flash_kernel(const unsigned short* __restrict__ qb,
                   const unsigned short* __restrict__ kb,
                   const unsigned short* __restrict__ vtb,
                   const float* __restrict__ cs,
                   const float* __restrict__ g,
                   const float* __restrict__ pm,
                   const float* __restrict__ tmax,
                   const float* __restrict__ norm_w,
                   float* __restrict__ out) {
    const int h        = blockIdx.y;
    const int row_base = blockIdx.x * BM_;
    const int warp     = threadIdx.x >> 5;
    const int lane     = threadIdx.x & 31;
    const int half     = lane >> 4;
    const int l16      = lane & 15;
    const int wrow     = row_base + warp * 16;

    // double-buffered async staging
    __shared__ __align__(16) unsigned short Kld[2][BN_ * DH_];   // [key][d]
    __shared__ __align__(16) unsigned short Vt[2][DH_ * BN_];    // [d][key]
    __shared__ __align__(16) unsigned short Pld[4][16 * BN_];    // per-warp [row][key]

    float m[8], acc[8], rowcs[8], pmrow[8], svec[8];
    v8f   H[8];
#pragma unroll
    for (int r = 0; r < 8; ++r) {
        const int t = wrow + r + 8 * half;
        m[r]     = -INFINITY;
        acc[r]   = 0.f;          // per-lane partial row-sum (reduced at end)
        rowcs[r] = cs[h * S_ + t];
        pmrow[r] = pm[h * S_ + t];
    }
#pragma unroll
    for (int n = 0; n < 8; ++n) H[n] = {};

    // Q A-fragments (already bf16, scale folded): row = l16, K contiguous 16*half
    v16bf qa[4];
    {
        const unsigned short* qp = qb + (size_t)(wrow + l16) * E_ + h * DH_ + 16 * half;
#pragma unroll
        for (int c = 0; c < 4; ++c) qa[c] = sfrag(qp + 32 * c);
    }

    // issue one tile (8 async b128 per thread) into LDS buffer `buf`
    auto issue_tile = [&](int kt, int buf) {
        const int j0 = kt * BN_;
#pragma unroll
        for (int rr = 0; rr < 4; ++rr) {
            const int c = threadIdx.x + rr * 128;          // 16B chunk id, 0..511
            const int krow = c >> 4, koff = (c & 15) * 8;  // 8 elems per chunk
            async_cp16(&kb[(size_t)(j0 + krow) * E_ + h * DH_ + koff], &Kld[buf][c * 8]);
            const int vd = c >> 2, voff = (c & 3) * 8;
            async_cp16(&vtb[(size_t)(h * DH_ + vd) * S_ + j0 + voff], &Vt[buf][c * 8]);
        }
    };

    const int ktmax = (row_base + BM_) / BN_;
    issue_tile(0, 0);
    for (int kt = 0; kt < ktmax; ++kt) {
        const int cur = kt & 1;
        const int j0  = kt * BN_;
        __syncthreads();  // all waves done reading buffer 1-cur (tile kt-1)
        if (kt + 1 < ktmax) {
            issue_tile(kt + 1, 1 - cur);  // prefetch next tile behind compute
            wait_async8();                // wait only for tile kt's 8 copies
        } else {
            wait_async0();
        }
        // per-lane gate values for this tile (L2-resident, coalesced)
        const float g0  = g[h * S_ + j0 + l16];
        const float g1  = g[h * S_ + j0 + 16 + l16];
        const float tmx = tmax[h * NT_ + kt];  // tile max of g (uniform)
        __syncthreads();  // tile kt's async data visible block-wide

        // ---- GEMM1: P = Qs @ K^T (two 16-col subtiles) ----
        v8f P0 = {}, P1 = {};
#pragma unroll
        for (int c = 0; c < 4; ++c) {
            v16bf b0 = sfrag(&Kld[cur][l16 * DH_ + 32 * c + 16 * half]);
            v16bf b1 = sfrag(&Kld[cur][(16 + l16) * DH_ + 32 * c + 16 * half]);
            P0 = __builtin_amdgcn_wmma_f32_16x16x32_bf16(false, qa[c], false, b0,
                                                         (short)0, P0, false, false);
            P1 = __builtin_amdgcn_wmma_f32_16x16x32_bf16(false, qa[c], false, b1,
                                                         (short)0, P1, false, false);
        }

        // ---- decay weights + online rescale (row max is precomputed) ----
        const int jg0 = j0 + l16, jg1 = j0 + 16 + l16;
#pragma unroll
        for (int r = 0; r < 8; ++r) {
            const int t = wrow + r + 8 * half;
            // exact row max of log_D over this tile, without any reduction:
            const float gmx = (t >= j0 + BN_ - 1) ? tmx
                              : ((t >= j0) ? pmrow[r] : -INFINITY);
            const float tm = rowcs[r] + gmx;
            const float nm = fmaxf(m[r], tm);
            const float sc = __expf(m[r] - nm);   // 0 on first tile (m = -inf)
            m[r]    = nm;
            svec[r] = sc;
            const float w0 = (jg0 <= t) ? __expf(rowcs[r] + g0 - nm) : 0.f;
            const float w1 = (jg1 <= t) ? __expf(rowcs[r] + g1 - nm) : 0.f;
            const float c0 = P0[r] * w0;
            const float c1 = P1[r] * w1;
            acc[r] = acc[r] * sc + (c0 + c1);     // per-lane partial
            Pld[warp][(r + 8 * half) * BN_ + l16]      = f2bf(c0);
            Pld[warp][(r + 8 * half) * BN_ + 16 + l16] = f2bf(c1);
        }
#pragma unroll
        for (int n = 0; n < 8; ++n)
#pragma unroll
            for (int r = 0; r < 8; ++r) H[n][r] *= svec[r];
        __syncthreads();

        // ---- GEMM2: H += P @ V ----
        v16bf pa = sfrag(&Pld[warp][l16 * BN_ + 16 * half]);
#pragma unroll
        for (int n = 0; n < 8; ++n) {
            v16bf vb = sfrag(&Vt[cur][(n * 16 + l16) * BN_ + 16 * half]);
            H[n] = __builtin_amdgcn_wmma_f32_16x16x32_bf16(false, pa, false, vb,
                                                           (short)0, H[n], false, false);
        }
    }

    // ---- finalize: normalizer (single deferred reduction) + group-norm ----
    float inv[8];
#pragma unroll
    for (int r = 0; r < 8; ++r) {
        const float rowsum = hsum16(acc[r]);
        inv[r] = 1.f / (fmaxf(fabsf(rowsum), __expf(-m[r])) + 1e-6f);
    }
    float s1[8], s2[8];
#pragma unroll
    for (int r = 0; r < 8; ++r) { s1[r] = 0.f; s2[r] = 0.f; }
#pragma unroll
    for (int n = 0; n < 8; ++n)
#pragma unroll
        for (int r = 0; r < 8; ++r) {
            const float x = H[n][r] * inv[r];
            H[n][r] = x;
            s1[r] += x;
            s2[r] += x * x;
        }
    float mean[8], rstd[8];
#pragma unroll
    for (int r = 0; r < 8; ++r) {
        const float a = hsum16(s1[r]) * (1.f / DH_);
        const float b = hsum16(s2[r]) * (1.f / DH_);
        mean[r] = a;
        rstd[r] = rsqrtf(fmaxf(b - a * a, 0.f) + 1e-5f);
    }
#pragma unroll
    for (int n = 0; n < 8; ++n) {
        const float wn = norm_w[h * DH_ + n * 16 + l16];
#pragma unroll
        for (int r = 0; r < 8; ++r) {
            const int t = wrow + r + 8 * half;
            out[(size_t)t * E_ + h * DH_ + n * 16 + l16] =
                (H[n][r] - mean[r]) * rstd[r] * wn;
        }
    }
}

extern "C" void kernel_launch(void* const* d_in, const int* in_sizes, int n_in,
                              void* d_out, int out_size, void* d_ws, size_t ws_size,
                              hipStream_t stream) {
    (void)in_sizes; (void)n_in; (void)out_size; (void)ws_size;
    const float* q    = (const float*)d_in[0];
    const float* k    = (const float*)d_in[1];
    const float* v    = (const float*)d_in[2];
    const float* igw  = (const float*)d_in[3];
    const float* igb  = (const float*)d_in[4];
    const float* fgw  = (const float*)d_in[5];
    const float* fgb  = (const float*)d_in[6];
    const float* nw   = (const float*)d_in[7];
    float* out = (float*)d_out;

    float* ig   = (float*)d_ws;                     // NH*S f32
    float* lsig = ig + NH_ * S_;                    // NH*S f32
    float* cs   = lsig + NH_ * S_;                  // NH*S f32
    float* g    = cs + NH_ * S_;                    // NH*S f32
    float* pm   = g + NH_ * S_;                     // NH*S f32
    float* tmax = pm + NH_ * S_;                    // NH*NT f32
    unsigned short* qb  = (unsigned short*)(tmax + NH_ * NT_ + 8);  // S*E bf16
    unsigned short* kb  = qb + (size_t)S_ * E_;                     // S*E bf16
    unsigned short* vtb = kb + (size_t)S_ * E_;                     // NH*DH*S bf16

    mlstm_convert_kernel<<<(S_ * E_) / 256, 256, 0, stream>>>(q, k, v, qb, kb, vtb);
    mlstm_gates_kernel<<<S_, 256, 0, stream>>>(q, k, v, igw, igb, fgw, fgb, ig, lsig);
    mlstm_cumsum_kernel<<<NH_, 256, 0, stream>>>(lsig, ig, cs, g, pm, tmax);
    dim3 grid(S_ / BM_, NH_);
    mlstm_flash_kernel<<<grid, 128, 0, stream>>>(qb, kb, vtb, cs, g, pm, tmax, nw, out);
}
